// MSDeformableAttention_condition_90658169684433
// MI455X (gfx1250) — matrix operand order repaired and checked by reading, not compiled, not run on hardware
//
#include <hip/hip_runtime.h>
#include <hip/hip_bf16.h>

// ---------------- types for WMMA ----------------
typedef __attribute__((ext_vector_type(16))) _Float16 v16h;
typedef __attribute__((ext_vector_type(8)))  _Float16 h8;
typedef __attribute__((ext_vector_type(8)))  float    v8f;

// Problem constants (match setup_inputs)
#define BS   8
#define LQ   2048
#define CDIM 256
#define H2C  16      // heads
#define HC   16      // channels per head
#define LVL  4
#define PTS  4
#define LVTOT 8500   // 80*80+40*40+20*20+10*10

// -------------------------------------------------------------------------
// Transpose + convert fp32 (K,N) row-major -> f16 (N,K) row-major
// -------------------------------------------------------------------------
__global__ void tconv_kernel(const float* __restrict__ W, _Float16* __restrict__ WT,
                             int K, int N) {
    int i = blockIdx.x * blockDim.x + threadIdx.x;
    int total = K * N;
    if (i >= total) return;
    int k = i % K;
    int n = i / K;
    WT[i] = (_Float16)W[k * N + n];
}

// -------------------------------------------------------------------------
// Elementwise fp32 -> f16
// -------------------------------------------------------------------------
__global__ void cvt_f16_kernel(const float* __restrict__ in, _Float16* __restrict__ out, int n) {
    int i = blockIdx.x * blockDim.x + threadIdx.x;
    if (i < n) out[i] = (_Float16)in[i];
}

// -------------------------------------------------------------------------
// Build an ISA-layout v16h fragment from two contiguous 16B chunks.
// -------------------------------------------------------------------------
__device__ __forceinline__ v16h mk_frag(const _Float16* __restrict__ p0,
                                        const _Float16* __restrict__ p1) {
    h8 lo = *(const h8*)p0;
    h8 hi = *(const h8*)p1;
    v16h r;
#pragma unroll
    for (int i = 0; i < 8; ++i) { r[i] = lo[i]; r[8 + i] = hi[i]; }
    return r;
}

// -------------------------------------------------------------------------
// WMMA GEMM: C[M,N] = A[M,K] (f16 row-major) @ B (given as BT f16, N x K
// row-major) + bias[N], fp32 output.
// Register-blocked: one wave computes a 32(M) x 64(N) macro-tile =
// 2 A-fragments x 4 B-fragments -> 8 independent v_wmma per K-step
// (~24B of fragment traffic per WMMA vs 64B for a 1x1 tile).
// Single-buffered on purpose: explicit double-buffering of fragments was
// tried and caused accumulator spills to scratch inside the inner loop.
// M % 32 == 0, N % 64 == 0, K % 32 == 0. 256 threads = 8 waves / block.
// -------------------------------------------------------------------------
__global__ void wmma_gemm_kernel(const _Float16* __restrict__ A,
                                 const _Float16* __restrict__ BT,
                                 const float* __restrict__ bias,
                                 float* __restrict__ C,
                                 int M, int N, int K) {
    const int lane = threadIdx.x & 31;
    const int wave = blockIdx.x * (blockDim.x >> 5) + (threadIdx.x >> 5);
    const int Mt2 = M >> 5;          // 32-row macro-tiles
    const int Ng  = N >> 6;          // 64-col macro-tiles
    if (wave >= Mt2 * Ng) return;    // wave-uniform: EXEC all-ones inside

    const int mt2 = wave / Ng;
    const int ng  = wave % Ng;

    const int l15 = lane & 15;
    // ISA 16-bit A 16x32 fragment: lanes 0-15 hold K {kb..kb+7, kb+16..kb+23},
    // lanes 16-31 hold K {kb+8..kb+15, kb+24..kb+31}; row = M0 + (lane&15).
    const int khalfA = (lane >> 4) << 3;   // 0 or 8
    // ISA 16-bit B 32x16 fragment: lanes 0-15 hold K {kb..kb+7, kb+16..kb+23} for col N0+lane,
    // lanes 16-31 hold K {kb+8..kb+15, kb+24..kb+31} for col N0+(lane-16).
    const int kqB    = (lane >> 4) << 4;   // 0 or 16

    const _Float16* __restrict__ arow0 = A + (size_t)(mt2 * 32 + l15) * K;
    const _Float16* __restrict__ arow1 = arow0 + (size_t)16 * K;
    const _Float16* brow[4];
#pragma unroll
    for (int j = 0; j < 4; ++j)
        brow[j] = BT + (size_t)(ng * 64 + j * 16 + l15) * K;

    v8f acc[2][4] = {};
    for (int kb = 0; kb < K; kb += 32) {
        v16h a0 = mk_frag(arow0 + kb + khalfA, arow0 + kb + 16 + khalfA);
        v16h a1 = mk_frag(arow1 + kb + khalfA, arow1 + kb + 16 + khalfA);
        v16h b[4];
#pragma unroll
        for (int j = 0; j < 4; ++j)
            b[j] = mk_frag(brow[j] + kb + khalfA, brow[j] + kb + 16 + khalfA);
#pragma unroll
        for (int j = 0; j < 4; ++j) {
            acc[0][j] = __builtin_amdgcn_wmma_f32_16x16x32_f16(
                false, a0, false, b[j], (short)0, acc[0][j], false, false);
            acc[1][j] = __builtin_amdgcn_wmma_f32_16x16x32_f16(
                false, a1, false, b[j], (short)0, acc[1][j], false, false);
        }
    }

    // C/D layout: VGPR r -> lanes 0-15: (M0+r, N0+lane); lanes 16-31: (M0+8+r, N0+lane-16)
    const int rhalf = (lane >> 4) << 3;    // 0 or 8
#pragma unroll
    for (int mi = 0; mi < 2; ++mi) {
        const int rbase = mt2 * 32 + mi * 16 + rhalf;
#pragma unroll
        for (int j = 0; j < 4; ++j) {
            const int col = ng * 64 + j * 16 + l15;
            const float bb = bias[col];
#pragma unroll
            for (int r = 0; r < 8; ++r)
                C[(size_t)(rbase + r) * N + col] = acc[mi][j][r] + bb;
        }
    }
}

// -------------------------------------------------------------------------
// Softmax over rows of 16 contiguous floats (in place)
// -------------------------------------------------------------------------
__global__ void softmax16_kernel(float* __restrict__ aw, int rows) {
    int r = blockIdx.x * blockDim.x + threadIdx.x;
    if (r >= rows) return;
    float* p = aw + (size_t)r * 16;
    float v[16];
    float m = -1e30f;
#pragma unroll
    for (int i = 0; i < 16; ++i) { v[i] = p[i]; m = fmaxf(m, v[i]); }
    float s = 0.f;
#pragma unroll
    for (int i = 0; i < 16; ++i) { v[i] = __expf(v[i] - m); s += v[i]; }
    float inv = 1.f / s;
#pragma unroll
    for (int i = 0; i < 16; ++i) p[i] = v[i] * inv;
}

// -------------------------------------------------------------------------
// Deformable sampling + attention combine.
// 16 lanes (one per channel) handle one (b,q,h) group -> coalesced 64B gathers.
// v32 layout: (b, Lv, H2, HC) f32.  Output written directly as f16 row
// (b*Lq+q, h*16+ch) feeding the final WMMA GEMM.
// -------------------------------------------------------------------------
__global__ void sample_kernel(const float* __restrict__ v32,
                              const float* __restrict__ refp,   // (bs,Lq,L,2)
                              const float* __restrict__ off,    // (bs,Lq,256)
                              const float* __restrict__ aw,     // (bs,Lq,256) softmaxed
                              _Float16* __restrict__ outh) {    // (bs*Lq, 256) f16
    const int t  = blockIdx.x * blockDim.x + threadIdx.x;
    const int gi = t >> 4;            // (b,q,h) group index
    const int ch = t & 15;
    const int total = BS * LQ * H2C;
    if (gi >= total) return;

    const int h = gi & 15;
    const int q = (gi >> 4) & (LQ - 1);
    const int b = gi >> 15;           // gi / (16*2048)

    const int HH[LVL] = {80, 40, 20, 10};
    const int ST[LVL] = {0, 6400, 8000, 8400};

    const int bq = b * LQ + q;
    const float* op = off + (size_t)bq * 256 + h * 32;   // (l,p,2)
    const float* ap = aw  + (size_t)bq * 256 + h * 16;   // (l,p)
    const float* rp = refp + (size_t)bq * (LVL * 2);

    const size_t vb = (size_t)b * LVTOT * CDIM + (size_t)h * HC + ch;

    float acc = 0.f;
#pragma unroll
    for (int l = 0; l < LVL; ++l) {
        const int H = HH[l], W = HH[l];
        const float rx = rp[l * 2 + 0];
        const float ry = rp[l * 2 + 1];
        const size_t vl = vb + (size_t)ST[l] * CDIM;
#pragma unroll
        for (int p = 0; p < PTS; ++p) {
            const float ox = op[(l * PTS + p) * 2 + 0];
            const float oy = op[(l * PTS + p) * 2 + 1];
            // loc = ref + off/[W,H]; x = loc_x*W - 0.5
            const float x = rx * (float)W + ox - 0.5f;
            const float y = ry * (float)H + oy - 0.5f;
            const float x0f = floorf(x), y0f = floorf(y);
            const float fx = x - x0f, fy = y - y0f;
            const int x0 = (int)x0f, y0 = (int)y0f;
            float s = 0.f;
#pragma unroll
            for (int dx = 0; dx < 2; ++dx) {
#pragma unroll
                for (int dy = 0; dy < 2; ++dy) {
                    const int xi = x0 + dx, yi = y0 + dy;
                    if (xi >= 0 && xi < W && yi >= 0 && yi < H) {
                        const float wx = dx ? fx : (1.f - fx);
                        const float wy = dy ? fy : (1.f - fy);
                        s += wx * wy * v32[vl + (size_t)(yi * W + xi) * CDIM];
                    }
                }
            }
            acc += ap[l * PTS + p] * s;
        }
    }
    outh[(size_t)bq * 256 + h * HC + ch] = (_Float16)acc;
}

// -------------------------------------------------------------------------
// Launch
// Inputs: 0 query, 1 reference_points, 2 value, 3 value_spatial_shapes,
//         4 W_samp, 5 b_samp, 6 W_attn, 7 b_attn, 8 W_val, 9 b_val,
//         10 W_out, 11 b_out
// -------------------------------------------------------------------------
extern "C" void kernel_launch(void* const* d_in, const int* in_sizes, int n_in,
                              void* d_out, int out_size, void* d_ws, size_t ws_size,
                              hipStream_t stream) {
    const float* query = (const float*)d_in[0];
    const float* refp  = (const float*)d_in[1];
    const float* value = (const float*)d_in[2];
    const float* W_samp = (const float*)d_in[4];
    const float* b_samp = (const float*)d_in[5];
    const float* W_attn = (const float*)d_in[6];
    const float* b_attn = (const float*)d_in[7];
    const float* W_val  = (const float*)d_in[8];
    const float* b_val  = (const float*)d_in[9];
    const float* W_out  = (const float*)d_in[10];
    const float* b_out  = (const float*)d_in[11];
    float* out = (float*)d_out;

    const int Mv = BS * LVTOT;     // 68000
    const int Mq = BS * LQ;        // 16384
    const int K1 = CDIM;           // 256
    const int K2 = 2 * CDIM;       // 512
    const int N1 = CDIM;           // 256

    // ---- workspace carve-out (all 256B aligned) ----
    char* ws = (char*)d_ws;
    size_t cur = 0;
    auto carve = [&](size_t bytes) -> char* {
        char* p = ws + cur;
        cur += (bytes + 255) & ~(size_t)255;
        return p;
    };
    _Float16* value_h = (_Float16*)carve((size_t)Mv * K1 * 2);  // 34.8 MB
    _Float16* query_h = (_Float16*)carve((size_t)Mq * K2 * 2);  // 16.8 MB
    _Float16* WvT     = (_Float16*)carve((size_t)N1 * K1 * 2);
    _Float16* WsT     = (_Float16*)carve((size_t)N1 * K2 * 2);
    _Float16* WaT     = (_Float16*)carve((size_t)N1 * K2 * 2);
    _Float16* WoT     = (_Float16*)carve((size_t)N1 * K1 * 2);
    float*    v32     = (float*)carve((size_t)Mv * N1 * 4);     // 69.6 MB
    float*    off32   = (float*)carve((size_t)Mq * N1 * 4);     // 16.8 MB
    float*    aw32    = (float*)carve((size_t)Mq * N1 * 4);     // 16.8 MB
    _Float16* outh    = (_Float16*)carve((size_t)Mq * N1 * 2);  // 8.4 MB
    (void)ws_size;  // total ~157 MB

    const int TB = 256;

    // 1) weight transpose+convert
    tconv_kernel<<<(K1 * N1 + TB - 1) / TB, TB, 0, stream>>>(W_val,  WvT, K1, N1);
    tconv_kernel<<<(K2 * N1 + TB - 1) / TB, TB, 0, stream>>>(W_samp, WsT, K2, N1);
    tconv_kernel<<<(K2 * N1 + TB - 1) / TB, TB, 0, stream>>>(W_attn, WaT, K2, N1);
    tconv_kernel<<<(K1 * N1 + TB - 1) / TB, TB, 0, stream>>>(W_out,  WoT, K1, N1);

    // 2) activation convert
    cvt_f16_kernel<<<(Mv * K1 + TB - 1) / TB, TB, 0, stream>>>(value, value_h, Mv * K1);
    cvt_f16_kernel<<<(Mq * K2 + TB - 1) / TB, TB, 0, stream>>>(query, query_h, Mq * K2);

    // 3) WMMA GEMMs: 8 waves / block, one 32x64 macro-tile per wave
    auto gemm_blocks = [](int M, int N) { return ((M >> 5) * (N >> 6) + 7) / 8; };
    wmma_gemm_kernel<<<gemm_blocks(Mv, N1), TB, 0, stream>>>(value_h, WvT, b_val,  v32,   Mv, N1, K1);
    wmma_gemm_kernel<<<gemm_blocks(Mq, N1), TB, 0, stream>>>(query_h, WsT, b_samp, off32, Mq, N1, K2);
    wmma_gemm_kernel<<<gemm_blocks(Mq, N1), TB, 0, stream>>>(query_h, WaT, b_attn, aw32,  Mq, N1, K2);

    // 4) softmax over L*P=16 per (b,q,h)
    const int rows = Mq * H2C;  // 262144
    softmax16_kernel<<<(rows + TB - 1) / TB, TB, 0, stream>>>(aw32, rows);

    // 5) deformable sampling + combine -> f16 rows
    const int sthreads = rows * HC;  // 4,194,304
    sample_kernel<<<sthreads / TB, TB, 0, stream>>>(v32, refp, off32, aw32, outh);

    // 6) output projection -> d_out (fp32)
    wmma_gemm_kernel<<<gemm_blocks(Mq, N1), TB, 0, stream>>>(outh, WoT, b_out, out, Mq, N1, K1);

    (void)in_sizes; (void)n_in; (void)out_size;
}
